// FixedTemporalSpectralGNN_10642928959590
// MI455X (gfx1250) — matrix operand order repaired and checked by reading, not compiled, not run on hardware
//
#include <hip/hip_runtime.h>
#include <math.h>
#include <stdint.h>

// ---------------- problem constants ----------------
#define Bn    8
#define NPGn  4096
#define Nn    32768        // Bn*NPGn
#define En    262144
#define E2n   294912       // En + Nn  (self loops appended)
#define HIDn  256
#define KMAXn 64
#define DINn  64

typedef _Float16 h16;
typedef __attribute__((ext_vector_type(16))) _Float16 v16h_t;
typedef __attribute__((ext_vector_type(8)))  _Float16 v8h_t;
typedef __attribute__((ext_vector_type(8)))  float    v8f_t;

__device__ __forceinline__ float eluf(float x){ return x > 0.f ? x : (expf(x) - 1.f); }
// order-preserving float<->uint map for atomicMax-based segment max
__device__ __forceinline__ unsigned encF(float f){ unsigned u = __float_as_uint(f); return (u & 0x80000000u) ? ~u : (u | 0x80000000u); }
__device__ __forceinline__ float    decF(unsigned u){ return __uint_as_float((u & 0x80000000u) ? (u & 0x7FFFFFFFu) : ~u); }

// async 16-byte global -> LDS copy (gfx1250, tracked by ASYNCcnt)
__device__ __forceinline__ void async_g2l_b128(unsigned lds_off, const void* gaddr){
  asm volatile("global_load_async_to_lds_b128 %0, %1, off" :: "v"(lds_off), "v"(gaddr) : "memory");
}

// ---------------- utility kernels ----------------
__global__ void k_fill_f32(float* p, float v, long n){
  long i = (long)blockIdx.x * 256 + threadIdx.x; if (i < n) p[i] = v;
}
__global__ void k_fill_u32(unsigned* p, unsigned v, long n){
  long i = (long)blockIdx.x * 256 + threadIdx.x; if (i < n) p[i] = v;
}
__global__ void k_cvt_f16(const float* __restrict__ s, h16* __restrict__ d, long n){
  long i = (long)blockIdx.x * 256 + threadIdx.x; if (i < n) d[i] = (h16)s[i];
}
// d[r*cols+c] = s[r*stride + off + c]   (strided weight-slice convert, e.g. tp.W columns)
__global__ void k_cvt_f16_slice(const float* __restrict__ s, h16* __restrict__ d,
                                int rows, int cols, int stride, int off){
  long i = (long)blockIdx.x * 256 + threadIdx.x;
  long n = (long)rows * cols; if (i >= n) return;
  int r = (int)(i / cols), c = (int)(i % cols);
  d[i] = (h16)s[(size_t)r * stride + off + c];
}
// copy N x 256 block into h_comb (stride 768) at column offset
__global__ void k_copy256(const float* __restrict__ src, float* __restrict__ dst, int off){
  long i = (long)blockIdx.x * 256 + threadIdx.x;
  int n = (int)(i >> 8), c = (int)(i & 255);
  dst[(size_t)n * 768 + off + c] = src[i];
}

// ---------------- WMMA GEMM:  Y[MxNout] = Xh[MxK] * Wh[NoutxK]^T + bias ----------------
// grid = (Nout/64, ceil(M/128)); block = 256 (8 waves); wave => 16 rows x 64 cols (4 tiles)
// Tiles staged in LDS via GLOBAL_LOAD_ASYNC_TO_LDS_B128, double-buffered, ASYNCcnt-tracked.
// A fragment (16x32 f16): lanes 0-15 row=lm K 0-7/16-23, lanes 16-31 row=lm K 8-15/24-31
// B fragment (32x16 f16): lanes 0-15 col=lm K 0-7/16-23, lanes 16-31 col=lm K 8-15/24-31
#define SAh 40   // LDS row stride in halfs (80B: 16B-aligned, bank-conflict-free)
__global__ void k_wmma_gemm(const h16* __restrict__ Xh, const h16* __restrict__ Wh,
                            const float* __restrict__ bias, float* __restrict__ Y,
                            int M, int K, int Nout){
  __shared__ h16 sA[2][128 * SAh];
  __shared__ h16 sB[2][64 * SAh];
  int tid = threadIdx.x;
  int wave = tid >> 5, lane = tid & 31;
  int lm = lane & 15, hi = lane >> 4;
  int mbase = blockIdx.y * 128 + wave * 16;
  int nbase = blockIdx.x * 64;
  int KT = K >> 5;

  // per-stage async staging: A = 512 b128 xfers (2/thread), B = 256 (1/thread)
  int ar0 = (tid * 2) >> 2,      as0 = (tid * 2) & 3;
  int ar1 = (tid * 2 + 1) >> 2,  as1 = (tid * 2 + 1) & 3;
  int br  = tid >> 2,            bs  = tid & 3;
  int gr0 = blockIdx.y * 128 + ar0; if (gr0 >= M) gr0 = M - 1;   // clamp, mask at store
  int gr1 = blockIdx.y * 128 + ar1; if (gr1 >= M) gr1 = M - 1;
  const h16* srcA0 = Xh + (size_t)gr0 * K + as0 * 8;
  const h16* srcA1 = Xh + (size_t)gr1 * K + as1 * 8;
  const h16* srcB  = Wh + (size_t)(nbase + br) * K + bs * 8;
  unsigned dA0 = (unsigned)(uintptr_t)&sA[0][ar0 * SAh + as0 * 8];
  unsigned dA1 = (unsigned)(uintptr_t)&sA[0][ar1 * SAh + as1 * 8];
  unsigned dB  = (unsigned)(uintptr_t)&sB[0][br  * SAh + bs  * 8];
  unsigned bufStepA = (unsigned)(sizeof(h16) * 128 * SAh);
  unsigned bufStepB = (unsigned)(sizeof(h16) * 64 * SAh);

  // prologue: stage 0 into buffer 0
  async_g2l_b128(dA0, srcA0);
  async_g2l_b128(dA1, srcA1);
  async_g2l_b128(dB,  srcB);

  v8f_t acc[4] = {};
  for (int kt = 0; kt < KT; kt++){
    int buf = kt & 1;
    if (kt + 1 < KT){
      int k0n = (kt + 1) << 5;
      unsigned bo = (kt + 1) & 1 ? 1u : 0u;
      async_g2l_b128(dA0 + bo * bufStepA, srcA0 + k0n);
      async_g2l_b128(dA1 + bo * bufStepA, srcA1 + k0n);
      async_g2l_b128(dB  + bo * bufStepB, srcB  + k0n);
      asm volatile("s_wait_asynccnt 3" ::: "memory");   // oldest stage complete
    } else {
      asm volatile("s_wait_asynccnt 0" ::: "memory");
    }
    __syncthreads();

    v16h_t a;
    {
      const h16* ap = &sA[buf][(wave * 16 + lm) * SAh + hi * 8];
      v8h_t a0 = *(const v8h_t*)ap;
      v8h_t a1 = *(const v8h_t*)(ap + 16);
      #pragma unroll
      for (int j = 0; j < 8; j++){ a[j] = a0[j]; a[j + 8] = a1[j]; }
    }
    #pragma unroll
    for (int nt = 0; nt < 4; nt++){
      v16h_t b;
      {
        const h16* bp = &sB[buf][(nt * 16 + lm) * SAh + hi * 8];
        v8h_t b0 = *(const v8h_t*)bp;
        v8h_t b1 = *(const v8h_t*)(bp + 16);
        #pragma unroll
        for (int j = 0; j < 8; j++){ b[j] = b0[j]; b[j + 8] = b1[j]; }
      }
      acc[nt] = __builtin_amdgcn_wmma_f32_16x16x32_f16(false, a, false, b,
                                                       (short)0, acc[nt], false, false);
    }
    __syncthreads();   // all waves done reading buf before it is re-staged
  }

  #pragma unroll
  for (int nt = 0; nt < 4; nt++){
    int col = nbase + nt * 16 + lm;
    float bv = bias ? bias[col] : 0.f;
    #pragma unroll
    for (int v = 0; v < 8; v++){
      int row = mbase + v + hi * 8;
      if (row < M) Y[(size_t)row * Nout + col] = acc[nt][v] + bv;
    }
  }
}

// ---------------- LayerNorm (D in {256,512}) with fused residual / elu / mask ----------------
__global__ void k_layernorm(const float* __restrict__ y, const float* __restrict__ res,
                            const float* __restrict__ g, const float* __restrict__ bta,
                            const float* __restrict__ mask, float* __restrict__ out,
                            int D, int preElu, int postElu, long ostride, long ooff){
  int row = blockIdx.x, t = threadIdx.x;
  __shared__ float red[256];
  int per = D >> 8;
  float v[2];
  const float* yr = y + (size_t)row * D;
  float s = 0.f;
  for (int j = 0; j < per; j++){
    float vv = yr[t + j * 256];
    if (res) vv += res[(size_t)row * D + t + j * 256];
    if (preElu) vv = eluf(vv);
    v[j] = vv; s += vv;
  }
  red[t] = s; __syncthreads();
  for (int o = 128; o > 0; o >>= 1){ if (t < o) red[t] += red[t + o]; __syncthreads(); }
  float mu = red[0] / D; __syncthreads();
  s = 0.f;
  for (int j = 0; j < per; j++){ float d = v[j] - mu; s += d * d; }
  red[t] = s; __syncthreads();
  for (int o = 128; o > 0; o >>= 1){ if (t < o) red[t] += red[t + o]; __syncthreads(); }
  float inv = rsqrtf(red[0] / D + 1e-5f);
  float mk = mask ? mask[row] : 1.f;
  for (int j = 0; j < per; j++){
    int c = t + j * 256;
    float o_ = (v[j] - mu) * inv * g[c] + bta[c];
    if (postElu) o_ = eluf(o_);
    out[(size_t)row * ostride + ooff + c] = o_ * mk;
  }
}

// ---------------- spectral encoder (per-batch tiny attention -> filter gains f) ----------------
__global__ void k_spec_encoder(const float* eigvals, const int* eig_mask,
  const float* W1, const float* b1, const float* g, const float* bn,
  const float* W2, const float* b2,
  const float* Wq, const float* Wk, const float* Wv,
  const float* bq, const float* bk, const float* bv,
  const float* Wo, const float* bo,
  const float* fgW1, const float* fgb1, const float* fgW2, const float* fgb2,
  float* f_gain){
  int b = blockIdx.x, t = threadIdx.x;     // t < 64
  __shared__ float sk[64 * 16], sv[64 * 16], sm[64];
  float ev = eigvals[b * 64 + t];
  sm[t] = (eig_mask[b * 64 + t] != 0) ? 1.f : 0.f;
  float h1[16], r[16], ef[16], q[16];
  float mu = 0.f, var = 0.f;
  for (int j = 0; j < 16; j++){ h1[j] = ev * W1[j] + b1[j]; mu += h1[j]; }
  mu *= (1.f / 16.f);
  for (int j = 0; j < 16; j++){ float d = h1[j] - mu; var += d * d; }
  float inv = rsqrtf(var * (1.f / 16.f) + 1e-5f);
  for (int j = 0; j < 16; j++){ float z = (h1[j] - mu) * inv * g[j] + bn[j]; r[j] = z > 0.f ? z : 0.f; }
  for (int j = 0; j < 16; j++){ float a = b2[j]; for (int c = 0; c < 16; c++) a += W2[j * 16 + c] * r[c]; ef[j] = a; }
  for (int j = 0; j < 16; j++){
    float aq = bq[j], ak = bk[j], av = bv[j];
    for (int c = 0; c < 16; c++){ float e = ef[c]; aq += Wq[j*16+c]*e; ak += Wk[j*16+c]*e; av += Wv[j*16+c]*e; }
    q[j] = aq; sk[t * 16 + j] = ak; sv[t * 16 + j] = av;
  }
  __syncthreads();
  float ctx[16];
  const float scl = 0.35355339059327373f;  // 1/sqrt(8)
  for (int h = 0; h < 2; h++){
    float mxl = -3.4e38f;
    for (int kp = 0; kp < 64; kp++){
      float l = 0.f;
      for (int d2 = 0; d2 < 8; d2++) l += q[h*8+d2] * sk[kp*16 + h*8 + d2];
      l *= scl; if (sm[kp] == 0.f) l = -1e9f;
      mxl = fmaxf(mxl, l);
    }
    float se = 0.f, cacc[8] = {0,0,0,0,0,0,0,0};
    for (int kp = 0; kp < 64; kp++){
      float l = 0.f;
      for (int d2 = 0; d2 < 8; d2++) l += q[h*8+d2] * sk[kp*16 + h*8 + d2];
      l *= scl; if (sm[kp] == 0.f) l = -1e9f;
      float e = expf(l - mxl); se += e;
      for (int d2 = 0; d2 < 8; d2++) cacc[d2] += e * sv[kp*16 + h*8 + d2];
    }
    for (int d2 = 0; d2 < 8; d2++) ctx[h*8+d2] = cacc[d2] / se;
  }
  float o[16];
  for (int j = 0; j < 16; j++){ float a = bo[j]; for (int c = 0; c < 16; c++) a += Wo[j*16+c] * ctx[c]; o[j] = a; }
  float acc2 = fgb2[0];
  for (int i2 = 0; i2 < 32; i2++){
    float a = fgb1[i2];
    for (int c = 0; c < 16; c++) a += fgW1[i2*16+c] * o[c];
    a = a > 0.f ? a : 0.f;
    acc2 += fgW2[i2] * a;
  }
  float fsc = tanhf(acc2);
  f_gain[b * 64 + t] = (sm[t] != 0.f) ? fsc : 0.f;
}

// ---------------- spectral filter ----------------
// xf[b,k,d] = sum_n U[b,n,k] * x[b,n,d]   (LDS-tiled, one block per batch)
__global__ void k_xf(const float* __restrict__ U, const float* __restrict__ x, float* __restrict__ xf){
  int b = blockIdx.x, t = threadIdx.x;
  __shared__ float sU[64 * 64], sX[64 * 64];
  int k = t >> 2, dg = (t & 3) << 4;
  float acc[16];
  #pragma unroll
  for (int j = 0; j < 16; j++) acc[j] = 0.f;
  for (int n0 = 0; n0 < NPGn; n0 += 64){
    __syncthreads();
    for (int l = t; l < 4096; l += 256){
      int r = l >> 6, c = l & 63;
      size_t gi = ((size_t)(b * NPGn + n0 + r)) * 64 + c;
      sU[l] = U[gi]; sX[l] = x[gi];
    }
    __syncthreads();
    for (int nn = 0; nn < 64; nn++){
      float u = sU[nn * 64 + k];
      const float* xrow = &sX[nn * 64 + dg];
      #pragma unroll
      for (int j = 0; j < 16; j++) acc[j] += u * xrow[j];
    }
  }
  #pragma unroll
  for (int j = 0; j < 16; j++) xf[(size_t)b * 4096 + k * 64 + dg + j] = acc[j];
}
// xs[n,d] = sum_k U[n,k] * f[b,k] * xf[b,k,d]
__global__ void k_xs(const float* __restrict__ U, const float* __restrict__ fg,
                     const float* __restrict__ xf, float* __restrict__ xs){
  long i = (long)blockIdx.x * 256 + threadIdx.x;
  int n = (int)(i >> 6), d = (int)(i & 63), b = n >> 12;
  float a = 0.f;
  for (int k = 0; k < 64; k++)
    a += U[(size_t)n * 64 + k] * fg[b * 64 + k] * xf[(size_t)b * 4096 + k * 64 + d];
  xs[i] = a;
}

// ---------------- GAT pieces ----------------
__global__ void k_hist(const int* attr, int* cnt){
  long e = (long)blockIdx.x * 256 + threadIdx.x;
  if (e < En) atomicAdd(&cnt[attr[e]], 1);
}
// eevec[t][j] = row_t . We[j,:]   (t=0..2 edge types, t=3 mean row)
__global__ void k_eeproj(const float* tab, const float* We, const int* cnt, float* eevec){
  int t = blockIdx.x;
  __shared__ float row[64];
  if (threadIdx.x < 64){
    int c = threadIdx.x;
    float r;
    if (t < 3) r = tab[t * 64 + c];
    else r = (cnt[0]*tab[c] + cnt[1]*tab[64 + c] + cnt[2]*tab[128 + c]) * (1.f / (float)En);
    row[c] = r;
  }
  __syncthreads();
  int j = threadIdx.x;
  float a = 0.f;
  for (int c = 0; c < 64; c++) a += row[c] * We[(size_t)j * 64 + c];
  eevec[t * 256 + j] = a;
}
__global__ void k_edge_logit(const int* __restrict__ ei, const int* __restrict__ attr,
                             const float* __restrict__ xl, const float* __restrict__ xr,
                             const float* __restrict__ eevec, const float* __restrict__ att,
                             float* __restrict__ logit, unsigned* __restrict__ mx){
  long i = (long)blockIdx.x * 256 + threadIdx.x;
  if (i >= (long)E2n * 4) return;
  int e = (int)(i >> 2), h = (int)(i & 3);
  int s, d, t;
  if (e < En){ s = ei[e]; d = ei[En + e]; t = attr[e]; } else { s = d = e - En; t = 3; }
  const float* pl = xl + (size_t)s * 256 + h * 64;
  const float* pr = xr + (size_t)d * 256 + h * 64;
  const float* pe = eevec + t * 256 + h * 64;
  const float* pa = att + h * 64;
  float a = 0.f;
  for (int c = 0; c < 64; c++){
    float m = pl[c] + pr[c] + pe[c];
    m = m > 0.f ? m : 0.2f * m;         // leaky_relu(0.2)
    a += m * pa[c];
  }
  logit[i] = a;
  atomicMax(&mx[d * 4 + h], encF(a));
}
__global__ void k_edge_exp(const int* __restrict__ ei, const float* __restrict__ logit,
                           const unsigned* __restrict__ mx, float* __restrict__ el,
                           float* __restrict__ esum){
  long i = (long)blockIdx.x * 256 + threadIdx.x;
  if (i >= (long)E2n * 4) return;
  int e = (int)(i >> 2), h = (int)(i & 3);
  int d = (e < En) ? ei[En + e] : (e - En);
  float ev = expf(logit[i] - decF(mx[d * 4 + h]));
  el[i] = ev;
  atomicAdd(&esum[d * 4 + h], ev);
}
__global__ void k_edge_scatter(const int* __restrict__ ei, const float* __restrict__ xl,
                               const float* __restrict__ el, const float* __restrict__ esum,
                               float* __restrict__ hc){
  long i = (long)blockIdx.x * 256 + threadIdx.x;
  if (i >= (long)E2n * 4) return;
  int e = (int)(i >> 2), h = (int)(i & 3);
  int s, d;
  if (e < En){ s = ei[e]; d = ei[En + e]; } else { s = d = e - En; }
  float alpha = el[i] / (esum[d * 4 + h] + 1e-16f);
  const float* pl = xl + (size_t)s * 256 + h * 64;
  float* po = hc + (size_t)d * 320 + h * 64;
  for (int c = 0; c < 64; c++) atomicAdd(&po[c], alpha * pl[c]);
}
__global__ void k_bias2(float* hc, const float* b1, const float* b2){
  long i = (long)blockIdx.x * 256 + threadIdx.x;
  int n = (int)(i >> 8), c = (int)(i & 255);
  hc[(size_t)n * 320 + c] += b1[c] + b2[c];
}
// temb -> hc columns [256,320)
__global__ void k_temb(const float* __restrict__ x, const float* __restrict__ tpW,
                       const float* __restrict__ tpb, float* __restrict__ hc){
  long i = (long)blockIdx.x * 256 + threadIdx.x;
  int n = (int)(i >> 6), j = (int)(i & 63);
  hc[(size_t)n * 320 + 256 + j] =
      x[(size_t)n * 64 + 0] * tpW[j * 2 + 0] + x[(size_t)n * 64 + 1] * tpW[j * 2 + 1] + tpb[j];
}

// ---------------- temporal gather-add of table projections ----------------
__global__ void k_tgather(float* __restrict__ hpart, const float* __restrict__ proj,
                          const int* __restrict__ steps){
  long i = (long)blockIdx.x * 256 + threadIdx.x;
  int n = (int)(i >> 8), c = (int)(i & 255);
  int st = steps[n];
  float a = hpart[i];
  #pragma unroll
  for (int s = 0; s < 3; s++){
    int id = (st >> s) % 100;
    a += proj[(size_t)(s * 100 + id) * 256 + c];
  }
  hpart[i] = a;
}

// ---------------- heads ----------------
__global__ void k_scores(const float* __restrict__ zz2, const float* __restrict__ W3,
                         const float* __restrict__ b3, float* __restrict__ out){
  long n = (long)blockIdx.x * 256 + threadIdx.x;
  float a = 0.f;
  const float* r = zz2 + (size_t)n * 256;
  for (int c = 0; c < 256; c++) a += r[c] * W3[c];
  out[n] = a + b3[0];
}
__global__ void k_ge(const float* __restrict__ hcomb, float* __restrict__ ge){
  int b = blockIdx.x / 3, ch = blockIdx.x % 3;
  int c = ch * 256 + threadIdx.x;
  float s = 0.f;
  const float* base = hcomb + (size_t)b * NPGn * 768 + c;
  for (int i2 = 0; i2 < NPGn; i2++) s += base[(size_t)i2 * 768];
  ge[b * 768 + c] = s * (1.f / (float)NPGn);
}
__global__ void k_value(const float* __restrict__ ge, const float* __restrict__ W1,
                        const float* __restrict__ b1, const float* __restrict__ W2,
                        const float* __restrict__ b2, float* __restrict__ out){
  int b = blockIdx.x, t = threadIdx.x;
  __shared__ float red[256];
  const float* gr = ge + b * 768;
  const float* wr = W1 + (size_t)t * 768;
  float a = 0.f;
  for (int c = 0; c < 768; c++) a += gr[c] * wr[c];
  a = eluf(a + b1[t]);
  red[t] = a * W2[t]; __syncthreads();
  for (int o = 128; o > 0; o >>= 1){ if (t < o) red[t] += red[t + o]; __syncthreads(); }
  if (t == 0) out[b] = 1.f / (1.f + expf(-(red[0] + b2[0])));
}

// ---------------- host-side helpers ----------------
static inline int cdiv(long n, int d){ return (int)((n + d - 1) / d); }

static void cvtX(hipStream_t s, const float* src, h16* dst, long n){
  k_cvt_f16<<<cdiv(n, 256), 256, 0, s>>>(src, dst, n);
}
static void gemmW(hipStream_t s, h16* Wh, const h16* Xh, const float* W,
                  const float* bias, float* Y, int M, int K, int Nout){
  long nw = (long)Nout * K;
  k_cvt_f16<<<cdiv(nw, 256), 256, 0, s>>>(W, Wh, nw);
  dim3 g(Nout / 64, (unsigned)((M + 127) / 128));
  k_wmma_gemm<<<g, 256, 0, s>>>(Xh, Wh, bias, Y, M, K, Nout);
}
static void ln(hipStream_t s, const float* y, const float* res, const float* g, const float* b,
               const float* mask, float* out, int rows, int D, int preElu, int postElu,
               long ostride, long ooff){
  k_layernorm<<<rows, 256, 0, s>>>(y, res, g, b, mask, out, D, preElu, postElu, ostride, ooff);
}

extern "C" void kernel_launch(void* const* d_in, const int* in_sizes, int n_in,
                              void* d_out, int out_size, void* d_ws, size_t ws_size,
                              hipStream_t stream){
  (void)in_sizes; (void)n_in; (void)out_size; (void)ws_size;
  // ---- unpack inputs: setup_inputs() dict insertion order, params pytree leaves first ----
  int idx = 0;
  auto F = [&](){ return (const float*)d_in[idx++]; };
  auto I = [&](){ return (const int*)d_in[idx++]; };
  // params.sp
  const float *enc_W1=F(),*enc_b1=F(),*enc_g=F(),*enc_bn=F(),*enc_W2=F(),*enc_b2=F(),
              *Wq=F(),*Wk=F(),*Wv=F(),*bq=F(),*bk=F(),*bv=F(),*Wo=F(),*bo=F(),
              *fg_W1=F(),*fg_b1=F(),*fg_W2=F(),*fg_b2=F(),
              *out_W=F(),*out_b=F(),*out_g=F(),*out_bn=F();
  // params.st
  const float *edge_tab=F(),*in_W=F(),*in_b=F();
  struct Gat { const float *Wl,*bl,*Wr,*br,*We,*att,*bias; };
  struct Lay { Gat g1, g2; const float *tp_W,*tp_b,*cb_W,*cb_b,*n_g,*n_b; } lay[3];
  for (int l = 0; l < 3; l++){
    lay[l].g1 = Gat{F(),F(),F(),F(),F(),F(),F()};
    lay[l].g2 = Gat{F(),F(),F(),F(),F(),F(),F()};
    lay[l].tp_W=F(); lay[l].tp_b=F(); lay[l].cb_W=F(); lay[l].cb_b=F(); lay[l].n_g=F(); lay[l].n_b=F();
  }
  // params.tp
  const float* tabs[3] = {F(), F(), F()};
  const float *tpW=F(),*tpb=F(),*tpg=F(),*tpbn=F();
  // params.fs
  const float *fsW1=F(),*fsb1=F(),*fsg1=F(),*fsbn1=F(),*fsW2=F(),*fsb2=F(),
              *fsg2=F(),*fsbn2=F(),*fsW3=F(),*fsb3=F();
  // params.vh
  const float *vhW1=F(),*vhb1=F(),*vhW2=F(),*vhb2=F();
  // top-level tensors
  const float* x            = F();
  const int*   edge_index   = I();
  const float* derived_mask = F();
  const int*   step_numbers = I();
  const float* eigvecs      = F();
  const float* eigvals      = F();
  const int*   eig_mask     = I();
  const int*   edge_attr    = I();
  const int*   batch        = I(); (void)batch;   // contiguous: b = n / NPG

  // ---- workspace bump allocator ----
  char* wsb = (char*)d_ws; size_t off = 0;
  auto alloc = [&](size_t bytes)->void*{ void* p = wsb + off; off += (bytes + 255) & ~(size_t)255; return p; };
  float*    f_gain = (float*)   alloc((size_t)Bn * 64 * 4);
  float*    xf     = (float*)   alloc((size_t)Bn * 64 * 64 * 4);
  float*    xs     = (float*)   alloc((size_t)Nn * 64 * 4);
  float*    h_spec = (float*)   alloc((size_t)Nn * 256 * 4);
  float*    hA     = (float*)   alloc((size_t)Nn * 256 * 4);
  float*    hB     = (float*)   alloc((size_t)Nn * 256 * 4);
  float*    xl     = (float*)   alloc((size_t)Nn * 256 * 4);
  float*    xr     = (float*)   alloc((size_t)Nn * 256 * 4);   // reused later as zz2
  float*    hc     = (float*)   alloc((size_t)Nn * 320 * 4);
  float*    logit  = (float*)   alloc((size_t)E2n * 4 * 4);
  float*    el     = (float*)   alloc((size_t)E2n * 4 * 4);
  unsigned* mx     = (unsigned*)alloc((size_t)Nn * 4 * 4);
  float*    esum   = (float*)   alloc((size_t)Nn * 4 * 4);
  float*    eevec  = (float*)   alloc(4 * 256 * 4);
  int*      cnt    = (int*)     alloc(16 * 4);
  float*    proj   = (float*)   alloc((size_t)3 * 100 * 256 * 4);
  float*    hpart  = (float*)   alloc((size_t)Nn * 256 * 4);
  float*    zz1    = (float*)   alloc((size_t)Nn * 512 * 4);
  float*    ge     = (float*)   alloc((size_t)Bn * 768 * 4);
  h16*      Xh     = (h16*)     alloc((size_t)Nn * 768 * 2);
  h16*      Wh     = (h16*)     alloc((size_t)512 * 768 * 2);
  float*    zz2    = xr;

  float* outScores = (float*)d_out;
  float* hcomb     = (float*)d_out + Nn;                    // N x 768
  float* outValue  = (float*)d_out + Nn + (size_t)Nn * 768; // 8

  // ================= 1. spectral branch =================
  k_spec_encoder<<<Bn, 64, 0, stream>>>(eigvals, eig_mask,
      enc_W1, enc_b1, enc_g, enc_bn, enc_W2, enc_b2,
      Wq, Wk, Wv, bq, bk, bv, Wo, bo, fg_W1, fg_b1, fg_W2, fg_b2, f_gain);
  k_xf<<<Bn, 256, 0, stream>>>(eigvecs, x, xf);
  k_xs<<<cdiv((long)Nn * 64, 256), 256, 0, stream>>>(eigvecs, f_gain, xf, xs);
  cvtX(stream, xs, Xh, (long)Nn * 64);
  gemmW(stream, Wh, Xh, out_W, out_b, hB, Nn, 64, 256);         // hB = xs @ out_W^T + b
  ln(stream, hB, nullptr, out_g, out_bn, nullptr, h_spec, Nn, 256, 0, 0, 256, 0);
  k_copy256<<<Nn, 256, 0, stream>>>(h_spec, hcomb, 0);          // h_comb[:, 0:256]

  // ================= 2. spatial GNN =================
  cvtX(stream, x, Xh, (long)Nn * 64);
  gemmW(stream, Wh, Xh, in_W, in_b, hA, Nn, 64, 256);           // h0
  k_fill_u32<<<1, 16, 0, stream>>>((unsigned*)cnt, 0u, 16);
  k_hist<<<cdiv(En, 256), 256, 0, stream>>>(edge_attr, cnt);

  float* hcur = hA; float* hnext = hB;
  for (int i = 0; i < 3; i++){
    const Lay& L = lay[i];
    cvtX(stream, hcur, Xh, (long)Nn * 256);
    k_fill_f32<<<cdiv((long)Nn * 320, 256), 256, 0, stream>>>(hc, 0.f, (long)Nn * 320);
    const Gat* gs[2] = {&L.g1, &L.g2};
    for (int gi = 0; gi < 2; gi++){
      const Gat& G = *gs[gi];
      gemmW(stream, Wh, Xh, G.Wl, G.bl, xl, Nn, 256, 256);
      gemmW(stream, Wh, Xh, G.Wr, G.br, xr, Nn, 256, 256);
      k_eeproj<<<4, 256, 0, stream>>>(edge_tab, G.We, cnt, eevec);
      k_fill_u32<<<cdiv((long)Nn * 4, 256), 256, 0, stream>>>(mx, 0x007FFFFFu, (long)Nn * 4); // enc(-inf)
      k_fill_f32<<<cdiv((long)Nn * 4, 256), 256, 0, stream>>>(esum, 0.f, (long)Nn * 4);
      k_edge_logit  <<<E2n * 4 / 256, 256, 0, stream>>>(edge_index, edge_attr, xl, xr, eevec, G.att, logit, mx);
      k_edge_exp    <<<E2n * 4 / 256, 256, 0, stream>>>(edge_index, logit, mx, el, esum);
      k_edge_scatter<<<E2n * 4 / 256, 256, 0, stream>>>(edge_index, xl, el, esum, hc);
    }
    k_bias2<<<Nn, 256, 0, stream>>>(hc, L.g1.bias, L.g2.bias);
    k_temb<<<cdiv((long)Nn * 64, 256), 256, 0, stream>>>(x, L.tp_W, L.tp_b, hc);
    cvtX(stream, hc, Xh, (long)Nn * 320);
    gemmW(stream, Wh, Xh, L.cb_W, L.cb_b, xl, Nn, 320, 256);    // xl = combine output
    ln(stream, xl, hcur, L.n_g, L.n_b, nullptr, hnext, Nn, 256, 0, (i < 2) ? 1 : 0, 256, 0);
    float* t = hcur; hcur = hnext; hnext = t;
  }
  k_copy256<<<Nn, 256, 0, stream>>>(hcur, hcomb, 256);          // h_comb[:, 256:512] = h_spatial

  // ================= 3. temporal branch =================
  // lin(concat(embs,h_spec), tp.W) == sum_s proj_s[idx_s] + h_spec @ Wslice3^T + b
  for (int s3 = 0; s3 < 3; s3++){
    cvtX(stream, tabs[s3], Xh, 100L * 256);
    k_cvt_f16_slice<<<cdiv(256L * 256, 256), 256, 0, stream>>>(tpW, Wh, 256, 256, 1024, s3 * 256);
    dim3 g(4, 1);
    k_wmma_gemm<<<g, 256, 0, stream>>>(Xh, Wh, nullptr, proj + (size_t)s3 * 100 * 256, 100, 256, 256);
  }
  cvtX(stream, h_spec, Xh, (long)Nn * 256);
  k_cvt_f16_slice<<<cdiv(256L * 256, 256), 256, 0, stream>>>(tpW, Wh, 256, 256, 1024, 768);
  {
    dim3 g(4, (unsigned)(Nn / 128));
    k_wmma_gemm<<<g, 256, 0, stream>>>(Xh, Wh, tpb, hpart, Nn, 256, 256);
  }
  k_tgather<<<Nn, 256, 0, stream>>>(hpart, proj, step_numbers);
  // h_temp = LN(elu(pre)) * mask, written straight into h_comb[:, 512:768]
  ln(stream, hpart, nullptr, tpg, tpbn, derived_mask, hcomb, Nn, 256, 1, 0, 768, 512);

  // ================= 4. final scorer =================
  cvtX(stream, hcomb, Xh, (long)Nn * 768);
  gemmW(stream, Wh, Xh, fsW1, fsb1, zz1, Nn, 768, 512);
  ln(stream, zz1, nullptr, fsg1, fsbn1, nullptr, zz1, Nn, 512, 1, 0, 512, 0);
  cvtX(stream, zz1, Xh, (long)Nn * 512);
  gemmW(stream, Wh, Xh, fsW2, fsb2, zz2, Nn, 512, 256);
  ln(stream, zz2, nullptr, fsg2, fsbn2, nullptr, zz2, Nn, 256, 1, 0, 256, 0);
  k_scores<<<Nn / 256, 256, 0, stream>>>(zz2, fsW3, fsb3, outScores);

  // ================= 5. value head =================
  k_ge<<<Bn * 3, 256, 0, stream>>>(hcomb, ge);
  k_value<<<Bn, 256, 0, stream>>>(ge, vhW1, vhb1, vhW2, vhb2, outValue);
}
